// Transformer_75084618269175
// MI455X (gfx1250) — matrix operand order repaired and checked by reading, not compiled, not run on hardware
//
#include <hip/hip_runtime.h>

// ---------------------------------------------------------------------------
// DiT forward for gfx1250 (MI455X). All large GEMMs via v_wmma_f32_16x16x32_bf16,
// LDS-staged, double-buffered, with gfx1250 async global->LDS copies
// (global_load_async_to_lds_b128 / s_wait_asynccnt via inline asm).
// ---------------------------------------------------------------------------

typedef __attribute__((ext_vector_type(16))) __bf16 bf16x16;
typedef __attribute__((ext_vector_type(8)))  __bf16 bf16x8;
typedef __attribute__((ext_vector_type(8)))  float  floatx8;

#define LOG1E4 9.210340371976184f /* ln(10000) */

// Flip to 0 to fall back to plain load+ds_store staging.
#define USE_ASYNC_LDS 1

__device__ __forceinline__ floatx8 wmma_bf16(bf16x16 a, bf16x16 b, floatx8 c) {
  // (neg_a, A, neg_b, B, c_mod, C, reuse_a, reuse_b)
  return __builtin_amdgcn_wmma_f32_16x16x32_bf16(false, a, false, b, (short)0, c,
                                                 false, false);
}

// 16-byte global -> LDS copy. Async (ASYNCcnt) on gfx1250.
__device__ __forceinline__ void cp16_g2l(__bf16* l, const __bf16* g) {
#if USE_ASYNC_LDS
  // vDST = per-lane LDS byte address (low 32 bits of generic pointer),
  // vADDR = 64-bit global address, no SADDR.
  asm volatile("global_load_async_to_lds_b128 %0, %1, off"
               :
               : "v"((unsigned)(unsigned long long)l), "v"(g)
               : "memory");
#else
  *(uint4*)l = *(const uint4*)g;
#endif
}

__device__ __forceinline__ void wait_stage() {
#if USE_ASYNC_LDS
  asm volatile("s_wait_asynccnt 0" ::: "memory");
#endif
  __syncthreads();
}

// A fragment from LDS (row-major, 32 elems/row):
//   lanes 0-15 : row=lane,    elems 0-15 -> K 0-15
//   lanes 16-31: row=lane-16, elems 0-15 -> K 16-31
__device__ __forceinline__ bf16x16 ld_frag_a(const __bf16* p) {
  bf16x8 lo = *(const bf16x8*)p;
  bf16x8 hi = *(const bf16x8*)(p + 8);
  bf16x16 r;
#pragma unroll
  for (int e = 0; e < 8; ++e) { r[e] = lo[e]; r[8 + e] = hi[e]; }
  return r;
}

// ---------------------------------------------------------------------------
// Batched WMMA GEMM: C[M,N] = alpha * A[M,K] * B[K,N]
//  - A row-major bf16, lda
//  - B: TRANSB=0 -> row-major K x N; TRANSB=1 -> source is N x K row-major (B^T)
//  - batch z decomposed (outer, inner) with inner count batchH; per-operand
//    element offsets outer*xOut + inner*xIn.
// Block: 256 threads (8 waves, 4x2); block tile 128(M) x 64(N); K-step 32.
// LDS: A row-major 128x32, B transposed [n][k] 64x32; double buffered (24 KB).
// Requires M%128==0, N%64==0, K%32==0 (true for all call sites).
// ---------------------------------------------------------------------------
template <bool TRANSB, typename OUTT>
__global__ __launch_bounds__(256) void k_gemm(
    const __bf16* __restrict__ A, const __bf16* __restrict__ B,
    OUTT* __restrict__ C, int lda, int ldb, int ldc, int M, int N, int K,
    float alpha, int batchH, long long aOut, long long aIn, long long bOut,
    long long bIn, long long cOut, long long cIn) {
  __shared__ __bf16 As[2][128 * 32];
  __shared__ __bf16 Bs[2][64 * 32];  // [n][k]

  const int bat = blockIdx.z;
  const int bo = bat / batchH, bi = bat % batchH;
  const __bf16* Ab = A + bo * aOut + bi * aIn;
  const __bf16* Bb = B + bo * bOut + bi * bIn;
  OUTT* Cb = C + bo * cOut + bi * cIn;

  const int tid = threadIdx.x;
  const int lane = tid & 31;
  const int wave = tid >> 5;
  const int wm = (wave >> 1) * 32;  // wave row offset within block tile
  const int wn = (wave & 1) * 32;   // wave col offset within block tile
  const int m_blk = blockIdx.y * 128;
  const int n_blk = blockIdx.x * 64;

  // staging assignments
  const int sa_row = tid >> 1, sa_seg = (tid & 1) * 16;  // A: 16 elems/thread
  const __bf16* Ag = Ab + (long long)(m_blk + sa_row) * lda + sa_seg;
  __bf16* AsDst0 = &As[0][sa_row * 32 + sa_seg];
  __bf16* AsDst1 = &As[1][sa_row * 32 + sa_seg];

  auto stage = [&](int buf, int k0) {
    cp16_g2l(buf ? AsDst1 : AsDst0, Ag + k0);
    if (k0 + 64 < K) __builtin_prefetch(Ag + k0 + 64, 0, 0);
    if constexpr (TRANSB) {
      if (tid < 128) {
        const int n = tid >> 1, seg = (tid & 1) * 16;
        cp16_g2l(&Bs[buf][n * 32 + seg],
                 Bb + (long long)(n_blk + n) * ldb + k0 + seg);
      }
    } else {
      // transpose K x N -> [n][k] via LDS scatter
      const int kk = tid >> 3, ng = (tid & 7) * 8;
      const __bf16* src = Bb + (long long)(k0 + kk) * ldb + n_blk + ng;
      uint4 v = *(const uint4*)src;
      const __bf16* pv = (const __bf16*)&v;
#pragma unroll
      for (int j = 0; j < 8; ++j) Bs[buf][(ng + j) * 32 + kk] = pv[j];
    }
  };

  floatx8 z = {0.f, 0.f, 0.f, 0.f, 0.f, 0.f, 0.f, 0.f};
  floatx8 acc00 = z, acc01 = z, acc10 = z, acc11 = z;

  const int arow = lane & 15;
  const int akb = (lane < 16) ? 0 : 16;
  const int nb = lane & 15;
  const int bkb = (lane < 16) ? 0 : 16;

  const int nk = K >> 5;
  stage(0, 0);
  wait_stage();

  for (int ik = 0; ik < nk; ++ik) {
    const int buf = ik & 1;
    if (ik + 1 < nk) stage(buf ^ 1, (ik + 1) << 5);

    const __bf16* Asb = As[buf];
    const __bf16* Bsb = Bs[buf];
    bf16x16 b0 = *(const bf16x16*)(Bsb + (wn + nb) * 32 + bkb);
    bf16x16 b1 = *(const bf16x16*)(Bsb + (wn + 16 + nb) * 32 + bkb);
    bf16x16 a0 = ld_frag_a(Asb + (wm + arow) * 32 + akb);
    bf16x16 a1 = ld_frag_a(Asb + (wm + 16 + arow) * 32 + akb);

    acc00 = wmma_bf16(a0, b0, acc00);
    acc01 = wmma_bf16(a0, b1, acc01);
    acc10 = wmma_bf16(a1, b0, acc10);
    acc11 = wmma_bf16(a1, b1, acc11);

    if (ik + 1 < nk) wait_stage();
  }

  // C/D layout: VGPR r: lanes 0-15 -> row r, lanes 16-31 -> row 8+r; col = lane&15
  const int rb = (lane < 16) ? 0 : 8;
  const int cn0 = n_blk + wn + (lane & 15);
  const int cn1 = cn0 + 16;
  const int rm = m_blk + wm;
#pragma unroll
  for (int r = 0; r < 8; ++r) {
    Cb[(long long)(rm + rb + r) * ldc + cn0] = (OUTT)(acc00[r] * alpha);
    Cb[(long long)(rm + rb + r) * ldc + cn1] = (OUTT)(acc01[r] * alpha);
    Cb[(long long)(rm + 16 + rb + r) * ldc + cn0] = (OUTT)(acc10[r] * alpha);
    Cb[(long long)(rm + 16 + rb + r) * ldc + cn1] = (OUTT)(acc11[r] * alpha);
  }
}

// ---------------------------------------------------------------------------
// f32 -> bf16 cast (per-layer weight conversion), grid-stride
// ---------------------------------------------------------------------------
__global__ void k_cast_bf16(const float* __restrict__ s, __bf16* __restrict__ d,
                            long long n) {
  long long i = (long long)blockIdx.x * 256 + threadIdx.x;
  const long long stride = (long long)gridDim.x * 256;
  for (; i < n; i += stride) d[i] = (__bf16)s[i];
}

// ---------------------------------------------------------------------------
// Patch embed: x[B,32,32,4] -> xt[B,256,1024] f32, + 2D sincos pos-embed
// ---------------------------------------------------------------------------
__global__ void k_patch(const float* __restrict__ x, const float* __restrict__ pW,
                        float* __restrict__ xt) {
  long long idx = (long long)blockIdx.x * 256 + threadIdx.x;  // B*256*1024
  const int o = (int)(idx & 1023);
  const long long bn = idx >> 10;
  const int n = (int)(bn & 255);
  const int b = (int)(bn >> 8);
  const int ph = n >> 4, pw = n & 15;
  const float* xb = x + (long long)b * 32 * 32 * 4;
  float acc = 0.f;
#pragma unroll
  for (int i = 0; i < 2; ++i)
#pragma unroll
    for (int j = 0; j < 2; ++j)
#pragma unroll
      for (int cc = 0; cc < 4; ++cc) {
        const int k = (i * 2 + j) * 4 + cc;
        acc += xb[(((ph * 2 + i) * 32) + (pw * 2 + j)) * 4 + cc] * pW[k * 1024 + o];
      }
  // pos embed: [0:256] sin(w*om), [256:512] cos(w*om), [512:768] sin(h*om), [768:1024] cos(h*om)
  const int seg = o >> 8;
  const int jj = o & 255;
  const float pos = (seg < 2) ? (float)pw : (float)ph;
  const float om = __expf(-LOG1E4 * (float)jj / 256.0f);
  const float ang = pos * om;
  acc += ((seg & 1) == 0) ? sinf(ang) : cosf(ang);
  xt[idx] = acc;
}

// ---------------------------------------------------------------------------
// Conditioning stage 1: hidden[b,1024] = silu(temb[b,256] @ t_W1)
// ---------------------------------------------------------------------------
__global__ void k_cond1(const float* __restrict__ t, const float* __restrict__ W1,
                        float* __restrict__ hidden) {
  __shared__ float temb[256];
  const int b = blockIdx.x, tid = threadIdx.x;
  const float tv = t[b];
  const float fr = __expf(-LOG1E4 * (float)(tid & 127) / 128.0f);
  const float a = tv * fr;
  temb[tid] = ((tid < 128) ? cosf(a) : sinf(a)) * 1.4142135623730951f;
  __syncthreads();
  for (int j = 0; j < 4; ++j) {
    const int o = tid + j * 256;
    float acc = 0.f;
    for (int k = 0; k < 256; ++k) acc += temb[k] * W1[(long long)k * 1024 + o];
    hidden[(long long)b * 1024 + o] = acc / (1.0f + __expf(-acc));  // silu
  }
}

// ---------------------------------------------------------------------------
// Conditioning stage 2: sc[b,1024] = silu(hidden[b] @ t_W2 + y_table[y[b]])
// ---------------------------------------------------------------------------
__global__ void k_cond2(const float* __restrict__ hidden, const float* __restrict__ W2,
                        const float* __restrict__ ytab, const int* __restrict__ y,
                        float* __restrict__ sc) {
  const int b = blockIdx.x >> 2;
  const int o = ((blockIdx.x & 3) << 8) + threadIdx.x;
  const float* hb = hidden + (long long)b * 1024;
  float acc = 0.f;
  for (int k = 0; k < 1024; ++k) acc += hb[k] * W2[(long long)k * 1024 + o];
  acc += ytab[(long long)y[b] * 1024 + o];
  sc[(long long)b * 1024 + o] = acc / (1.0f + __expf(-acc));
}

// ---------------------------------------------------------------------------
// Small dense matmul (f32): out[b,o] = A[b,:] @ W[:,o] (+ bias), K x N
// ---------------------------------------------------------------------------
__global__ void k_smallmm(const float* __restrict__ A, const float* __restrict__ W,
                          const float* __restrict__ bias, float* __restrict__ out,
                          int K, int N) {
  const long long idx = (long long)blockIdx.x * 256 + threadIdx.x;
  const int o = (int)(idx % N);
  const int b = (int)(idx / N);
  const float* a = A + (long long)b * K;
  float acc = bias ? bias[o] : 0.f;
  for (int k = 0; k < K; ++k) acc += a[k] * W[(long long)k * N + o];
  out[idx] = acc;
}

// ---------------------------------------------------------------------------
// LayerNorm (eps 1e-6, population var) + modulate -> bf16.  S=256, h=1024.
// ---------------------------------------------------------------------------
__global__ void k_ln_mod(const float* __restrict__ x, const float* __restrict__ shift,
                         const float* __restrict__ scale, int cstride,
                         __bf16* __restrict__ out) {
  __shared__ float r1[256], r2[256];
  const int row = blockIdx.x;
  const int b = row >> 8;
  const int t = threadIdx.x;
  const float* xr = x + (long long)row * 1024;
  const float v0 = xr[t], v1 = xr[t + 256], v2 = xr[t + 512], v3 = xr[t + 768];
  r1[t] = v0 + v1 + v2 + v3;
  r2[t] = v0 * v0 + v1 * v1 + v2 * v2 + v3 * v3;
  __syncthreads();
  for (int st = 128; st > 0; st >>= 1) {
    if (t < st) { r1[t] += r1[t + st]; r2[t] += r2[t + st]; }
    __syncthreads();
  }
  const float mean = r1[0] * (1.0f / 1024.0f);
  const float var = r2[0] * (1.0f / 1024.0f) - mean * mean;
  const float rstd = rsqrtf(var + 1e-6f);
  const float* sh = shift + (long long)b * cstride;
  const float* sc = scale + (long long)b * cstride;
  __bf16* orow = out + (long long)row * 1024;
  orow[t]       = (__bf16)((v0 - mean) * rstd * sc[t]       + sh[t]);
  orow[t + 256] = (__bf16)((v1 - mean) * rstd * sc[t + 256] + sh[t + 256]);
  orow[t + 512] = (__bf16)((v2 - mean) * rstd * sc[t + 512] + sh[t + 512]);
  orow[t + 768] = (__bf16)((v3 - mean) * rstd * sc[t + 768] + sh[t + 768]);
}

// ---------------------------------------------------------------------------
// Row softmax over 256 cols; writes bf16 in place at the same row byte offset
// (f32 row r at byte r*1024 -> bf16 row r at element offset r*512).
// ---------------------------------------------------------------------------
__global__ void k_softmax(float* __restrict__ scores) {
  __shared__ float red[256];
  const long long row = blockIdx.x;
  float* rp = scores + row * 256;
  const int t = threadIdx.x;
  const float v = rp[t];
  red[t] = v;
  __syncthreads();
  for (int s = 128; s > 0; s >>= 1) {
    if (t < s) red[t] = fmaxf(red[t], red[t + s]);
    __syncthreads();
  }
  const float mx = red[0];
  __syncthreads();
  const float e = __expf(v - mx);
  red[t] = e;
  __syncthreads();
  for (int s = 128; s > 0; s >>= 1) {
    if (t < s) red[t] += red[t + s];
    __syncthreads();
  }
  const float inv = 1.0f / red[0];
  __syncthreads();  // everyone has read its f32 before bf16 overwrite
  ((__bf16*)rp)[t] = (__bf16)(e * inv);
}

// ---------------------------------------------------------------------------
// x += gate[b,:] * tmp  (adaLN-Zero residual)
// ---------------------------------------------------------------------------
__global__ void k_resgate(float* __restrict__ x, const float* __restrict__ tmp,
                          const float* __restrict__ gate, int cstride) {
  const long long i = (long long)blockIdx.x * 256 + threadIdx.x;
  const int o = (int)(i & 1023);
  const int b = (int)(i >> 18);  // i >> 10 = row; row >> 8 = b
  x[i] += gate[(long long)b * cstride + o] * tmp[i];
}

// ---------------------------------------------------------------------------
// tanh-approx GELU, f32 -> bf16
// ---------------------------------------------------------------------------
__global__ void k_gelu(const float* __restrict__ in, __bf16* __restrict__ out,
                       long long n) {
  long long i = (long long)blockIdx.x * 256 + threadIdx.x;
  const long long stride = (long long)gridDim.x * 256;
  for (; i < n; i += stride) {
    const float v = in[i];
    const float u = 0.7978845608028654f * (v + 0.044715f * v * v * v);
    out[i] = (__bf16)(0.5f * v * (1.0f + tanhf(u)));
  }
}

// ---------------------------------------------------------------------------
// Output head: xo = xm @ out_W + out_b, fused with unpatchify -> d_out[B,32,32,4]
// ---------------------------------------------------------------------------
__global__ void k_outproj(const __bf16* __restrict__ xm, const float* __restrict__ Wo,
                          const float* __restrict__ bo, float* __restrict__ out) {
  const long long idx = (long long)blockIdx.x * 256 + threadIdx.x;
  const int kk = (int)(idx & 15);
  const long long bn = idx >> 4;
  const int n = (int)(bn & 255);
  const int b = (int)(bn >> 8);
  const __bf16* xr = xm + bn * 1024;
  float acc = bo[kk];
  for (int k = 0; k < 1024; ++k) acc += (float)xr[k] * Wo[k * 16 + kk];
  const int p = kk >> 3, q = (kk >> 2) & 1, cc = kk & 3;
  const int ph = n >> 4, pw = n & 15;
  const int row = ph * 2 + p, col = pw * 2 + q;
  out[((((long long)b * 32 + row) * 32) + col) * 4 + cc] = acc;
}

// ---------------------------------------------------------------------------
// Orchestration
// ---------------------------------------------------------------------------
extern "C" void kernel_launch(void* const* d_in, const int* in_sizes, int n_in,
                              void* d_out, int out_size, void* d_ws, size_t ws_size,
                              hipStream_t stream) {
  (void)in_sizes; (void)n_in; (void)out_size; (void)ws_size;

  const float* in_x   = (const float*)d_in[0];
  const float* in_t   = (const float*)d_in[1];
  const int*   in_y   = (const int*)d_in[2];
  const float* patchW = (const float*)d_in[3];
  const float* ytab   = (const float*)d_in[4];
  const float* tW1    = (const float*)d_in[5];
  const float* tW2    = (const float*)d_in[6];
  const float* adalnW = (const float*)d_in[7];
  const float* qkvW   = (const float*)d_in[8];
  const float* projW  = (const float*)d_in[9];
  const float* mlp1W  = (const float*)d_in[10];
  const float* mlp2W  = (const float*)d_in[11];
  const float* omW    = (const float*)d_in[12];
  const float* omB    = (const float*)d_in[13];
  const float* outW   = (const float*)d_in[14];
  const float* outB   = (const float*)d_in[15];

  // workspace carve-up
  size_t off = 0;
  auto carve = [&](size_t bytes) -> void* {
    void* p = (char*)d_ws + off;
    off += (bytes + 255) & ~(size_t)255;
    return p;
  };
  float*  sc     = (float*)carve(32 * 1024 * 4);              // silu(c)
  float*  hidden = (float*)carve(32 * 1024 * 4);
  float*  cm     = (float*)carve(32 * 6144 * 4);
  float*  cmout  = (float*)carve(32 * 2048 * 4);
  float*  xres   = (float*)carve((size_t)8192 * 1024 * 4);    // residual stream
  __bf16* xm     = (__bf16*)carve((size_t)8192 * 1024 * 2);   // modulated acts
  __bf16* qkvb   = (__bf16*)carve((size_t)8192 * 3072 * 2);   // k|q|v
  float*  scores = (float*)carve((size_t)512 * 256 * 256 * 4);
  __bf16* yb     = (__bf16*)carve((size_t)8192 * 1024 * 2);   // attn out
  float*  tmp    = (float*)carve((size_t)8192 * 4096 * 4);    // gemm f32 out
  __bf16* h1     = (__bf16*)carve((size_t)8192 * 4096 * 2);   // gelu(mlp1)
  __bf16* wq     = (__bf16*)carve((size_t)1024 * 3072 * 2);
  __bf16* wp     = (__bf16*)carve((size_t)1024 * 1024 * 2);
  __bf16* w1     = (__bf16*)carve((size_t)1024 * 4096 * 2);
  __bf16* w2     = (__bf16*)carve((size_t)4096 * 1024 * 2);

  // 1) patch embed + pos embed -> residual stream
  k_patch<<<32768, 256, 0, stream>>>(in_x, patchW, xres);

  // 2) conditioning: sc = silu(te + y_table[y])
  k_cond1<<<32, 256, 0, stream>>>(in_t, tW1, hidden);
  k_cond2<<<128, 256, 0, stream>>>(hidden, tW2, ytab, in_y, sc);

  for (int i = 0; i < 12; ++i) {
    // per-layer bf16 weight conversion
    k_cast_bf16<<<4096, 256, 0, stream>>>(qkvW  + (long long)i * 1024 * 3072, wq, (long long)1024 * 3072);
    k_cast_bf16<<<4096, 256, 0, stream>>>(projW + (long long)i * 1024 * 1024, wp, (long long)1024 * 1024);
    k_cast_bf16<<<4096, 256, 0, stream>>>(mlp1W + (long long)i * 1024 * 4096, w1, (long long)1024 * 4096);
    k_cast_bf16<<<4096, 256, 0, stream>>>(mlp2W + (long long)i * 4096 * 1024, w2, (long long)4096 * 1024);

    // adaLN: cm[b, 6144] = sc[b] @ adaln_W[i]
    k_smallmm<<<768, 256, 0, stream>>>(sc, adalnW + (long long)i * 1024 * 6144,
                                       nullptr, cm, 1024, 6144);

    // LN + modulate (shift_msa, scale_msa)
    k_ln_mod<<<8192, 256, 0, stream>>>(xres, cm + 0, cm + 1024, 6144, xm);

    // qkv = xm @ Wqkv  (M=8192,K=1024,N=3072) -> bf16
    k_gemm<false, __bf16><<<dim3(48, 64, 1), 256, 0, stream>>>(
        xm, wq, qkvb, 1024, 3072, 3072, 8192, 3072, 1024, 1.0f,
        1, 0, 0, 0, 0, 0, 0);

    // scores[b,h] = (1/8) * Q @ K^T   (batched 512; TRANSB; f32 out)
    k_gemm<true, float><<<dim3(4, 2, 512), 256, 0, stream>>>(
        qkvb + 1024 /*q*/, qkvb + 0 /*k*/, scores, 3072, 3072, 256,
        256, 256, 64, 0.125f, 16,
        (long long)256 * 3072, 64,            // A: per-b, per-h
        (long long)256 * 3072, 64,            // B: per-b, per-h
        (long long)16 * 65536, 65536);        // C: per-b, per-h

    // softmax rows (in-place f32 -> bf16, bf16 row stride 512 elements)
    k_softmax<<<131072, 256, 0, stream>>>(scores);

    // y[b,:,h*64:...] = P @ V   (batched 512; bf16 out)
    k_gemm<false, __bf16><<<dim3(1, 2, 512), 256, 0, stream>>>(
        (const __bf16*)scores, qkvb + 2048 /*v*/, yb, 512, 3072, 1024,
        256, 64, 256, 1.0f, 16,
        (long long)16 * 131072, 131072,       // A: per-b, per-h (bf16 view)
        (long long)256 * 3072, 64,            // B
        (long long)256 * 1024, 64);           // C

    // proj: tmp = y @ Wp  (f32)
    k_gemm<false, float><<<dim3(16, 64, 1), 256, 0, stream>>>(
        yb, wp, tmp, 1024, 1024, 1024, 8192, 1024, 1024, 1.0f,
        1, 0, 0, 0, 0, 0, 0);
    // x += g_msa * tmp
    k_resgate<<<32768, 256, 0, stream>>>(xres, tmp, cm + 2048, 6144);

    // LN + modulate (shift_mlp, scale_mlp)
    k_ln_mod<<<8192, 256, 0, stream>>>(xres, cm + 3072, cm + 4096, 6144, xm);

    // mlp1: tmp = xm @ W1  (N=4096, f32)
    k_gemm<false, float><<<dim3(64, 64, 1), 256, 0, stream>>>(
        xm, w1, tmp, 1024, 4096, 4096, 8192, 4096, 1024, 1.0f,
        1, 0, 0, 0, 0, 0, 0);
    // gelu -> bf16
    k_gelu<<<16384, 256, 0, stream>>>(tmp, h1, (long long)8192 * 4096);
    // mlp2: tmp = h1 @ W2  (K=4096, f32)
    k_gemm<false, float><<<dim3(16, 64, 1), 256, 0, stream>>>(
        h1, w2, tmp, 4096, 1024, 1024, 8192, 1024, 4096, 1.0f,
        1, 0, 0, 0, 0, 0, 0);
    // x += g_mlp * tmp
    k_resgate<<<32768, 256, 0, stream>>>(xres, tmp, cm + 5120, 6144);
  }

  // output modulation: cmout = sc @ out_mod_W + out_mod_b  (N=2048)
  k_smallmm<<<256, 256, 0, stream>>>(sc, omW, omB, cmout, 1024, 2048);
  // final LN + modulate
  k_ln_mod<<<8192, 256, 0, stream>>>(xres, cmout + 0, cmout + 1024, 2048, xm);
  // head + unpatchify
  k_outproj<<<512, 256, 0, stream>>>(xm, outW, outB, (float*)d_out);
}
